// SupplyChainGNN_30502857736434
// MI455X (gfx1250) — matrix operand order repaired and checked by reading, not compiled, not run on hardware
//
#include <hip/hip_runtime.h>

typedef __attribute__((ext_vector_type(16))) _Float16 v16h;
typedef __attribute__((ext_vector_type(8)))  float    v8f;

#define HDIM  64
#define LDSS  68      // padded LDS row stride (floats)
#define WAVES 8
#define TPB   (WAVES * 32)

static __device__ __forceinline__ v8f wmma16(v16h a, v16h b, v8f c) {
  // D = A(16x32 f16) x B(32x16 f16) + C(16x16 f32)
  return __builtin_amdgcn_wmma_f32_16x16x32_f16(false, a, false, b, (short)0, c, false, false);
}

static __device__ __forceinline__ v16h pack16(float4 u0, float4 u1, float4 u2, float4 u3) {
  v16h a;
  a[0]  = (_Float16)u0.x; a[1]  = (_Float16)u0.y; a[2]  = (_Float16)u0.z; a[3]  = (_Float16)u0.w;
  a[4]  = (_Float16)u1.x; a[5]  = (_Float16)u1.y; a[6]  = (_Float16)u1.z; a[7]  = (_Float16)u1.w;
  a[8]  = (_Float16)u2.x; a[9]  = (_Float16)u2.y; a[10] = (_Float16)u2.z; a[11] = (_Float16)u2.w;
  a[12] = (_Float16)u3.x; a[13] = (_Float16)u3.y; a[14] = (_Float16)u3.z; a[15] = (_Float16)u3.w;
  return a;
}

// A fragment, row base p, lda==64, contiguous: halves[0..7]=K(k0+8hi..+7), [8..15]=K(k0+16+8hi..+7)
static __device__ __forceinline__ v16h load_a64(const float* p, int k0, int hi) {
  const float4* q1 = (const float4*)(p + k0 + 8 * hi);
  const float4* q2 = (const float4*)(p + k0 + 16 + 8 * hi);
  return pack16(q1[0], q1[1], q2[0], q2[1]);
}

// A fragment from LDS tile (row stride LDSS floats, 16B-aligned rows)
static __device__ __forceinline__ v16h load_a_lds(const float* base, int row, int k0, int hi) {
  const float* q = base + row * LDSS + k0 + 8 * hi;
  const float4* q1 = (const float4*)q;
  const float4* q2 = (const float4*)(q + 16);
  return pack16(q1[0], q1[1], q2[0], q2[1]);
}

// B fragment: lane holds col, halves[0..7]=K(k0+8hi..+7), [8..15]=K(k0+16+8hi..+7); clamped address + value select (no branches)
static __device__ __forceinline__ v16h load_bfrag(const float* __restrict__ W,
                                                  int k0, int col, int hi, int kmax) {
  v16h b;
#pragma unroll
  for (int m = 0; m < 16; ++m) {
    int k = k0 + 8 * hi + (m < 8 ? m : m + 8);
    int kc = (k < kmax) ? k : (kmax - 1);
    float v = W[kc * HDIM + col];
    b[m] = (_Float16)((k < kmax) ? v : 0.0f);
  }
  return b;
}

// ================= utility kernels =================
__global__ void k_zero(float* p, long n) {
  long i = (long)blockIdx.x * blockDim.x + threadIdx.x;
  if (i < n) p[i] = 0.0f;
}

__global__ void k_zero4(float4* p, long n4) {
  long i = (long)blockIdx.x * blockDim.x + threadIdx.x;
  if (i < n4) p[i] = make_float4(0.f, 0.f, 0.f, 0.f);
}

__global__ void k_deg(const int* __restrict__ src, const int* __restrict__ dst,
                      float* deg, float* scnt, long E) {
  long e = (long)blockIdx.x * blockDim.x + threadIdx.x;
  if (e >= E) return;
  atomicAdd(&deg[dst[e]], 1.0f);
  atomicAdd(&scnt[src[e]], 1.0f);
}

__global__ void k_dinv(const float* __restrict__ deg, float* dinv, int N) {
  int i = blockIdx.x * blockDim.x + threadIdx.x;
  if (i < N) dinv[i] = rsqrtf(deg[i] + 1.0f);
}

// ================= node encoder: relu(x@Wn1+bn1)@Wn2+bn2 =================
__global__ void k_node_enc(const float* __restrict__ x,
                           const float* __restrict__ Wn1, const float* __restrict__ bn1,
                           const float* __restrict__ Wn2, const float* __restrict__ bn2,
                           float* __restrict__ out, int N) {
  __shared__ float lds[WAVES][16 * LDSS];
  int wv = threadIdx.x >> 5, lane = threadIdx.x & 31;
  int hi = lane >> 4, l15 = lane & 15;
  int ntiles = (N + 15) >> 4;
  float* L = lds[wv];

  // loop-invariant: weight fragments + per-lane biases
  v16h w1[4], w2[2][4];
  float b1c[4], b2c[4];
#pragma unroll
  for (int t = 0; t < 4; ++t) {
    int col = t * 16 + l15;
    w1[t] = load_bfrag(Wn1, 0, col, hi, 5);
    w2[0][t] = load_bfrag(Wn2, 0, col, hi, HDIM);
    w2[1][t] = load_bfrag(Wn2, 32, col, hi, HDIM);
    b1c[t] = bn1[col];
    b2c[t] = bn2[col];
  }

  int stride = gridDim.x * WAVES;
  for (int tile = blockIdx.x * WAVES + wv; tile < ntiles; tile += stride) {
    long rowg = (long)tile * 16 + l15;
    if (rowg >= N) rowg = N - 1;
    const float* p = x + rowg * 5;

    // layer 1: [16,5] @ [5,64]   (A: only hi==0, t<5 nonzero; value-selected, no branches)
    v16h a;
#pragma unroll
    for (int t = 0; t < 8; ++t) {
      float v = (t < 5) ? p[t < 5 ? t : 0] : 0.0f;
      a[t]     = (_Float16)((hi == 0 && t < 5) ? v : 0.0f);
      a[8 + t] = (_Float16)0.0f;
    }
    v8f acc[4] = {};
#pragma unroll
    for (int t = 0; t < 4; ++t) acc[t] = wmma16(a, w1[t], acc[t]);

#pragma unroll
    for (int t = 0; t < 4; ++t) {
      int col = t * 16 + l15;
#pragma unroll
      for (int r = 0; r < 8; ++r)
        L[(8 * hi + r) * LDSS + col] = fmaxf(acc[t][r] + b1c[t], 0.0f);
    }
    __builtin_amdgcn_wave_barrier();

    // layer 2: [16,64] @ [64,64]
    v8f acc2[4] = {};
#pragma unroll
    for (int ks = 0; ks < 2; ++ks) {
      v16h a2 = load_a_lds(L, l15, ks * 32, hi);
#pragma unroll
      for (int t = 0; t < 4; ++t) acc2[t] = wmma16(a2, w2[ks][t], acc2[t]);
    }
    __builtin_amdgcn_wave_barrier();

#pragma unroll
    for (int r = 0; r < 8; ++r) {
      long gr = (long)tile * 16 + 8 * hi + r;
      if (gr < N) {
#pragma unroll
        for (int t = 0; t < 4; ++t)
          out[gr * HDIM + t * 16 + l15] = acc2[t][r] + b2c[t];
      }
    }
  }
}

// ================= plain GEMM: out[N,64] = A[N,64] @ W[64,64] =================
__global__ void k_gemm64(const float* __restrict__ A, const float* __restrict__ W,
                         float* __restrict__ out, int N) {
  int wv = threadIdx.x >> 5, lane = threadIdx.x & 31;
  int hi = lane >> 4, l15 = lane & 15;
  int ntiles = (N + 15) >> 4;

  v16h wf[2][4];
#pragma unroll
  for (int ks = 0; ks < 2; ++ks)
#pragma unroll
    for (int t = 0; t < 4; ++t)
      wf[ks][t] = load_bfrag(W, ks * 32, t * 16 + l15, hi, HDIM);

  int stride = gridDim.x * WAVES;
  for (int tile = blockIdx.x * WAVES + wv; tile < ntiles; tile += stride) {
    long rowg = (long)tile * 16 + l15;
    if (rowg >= N) rowg = N - 1;
    const float* p = A + rowg * HDIM;
    v8f acc[4] = {};
#pragma unroll
    for (int ks = 0; ks < 2; ++ks) {
      v16h a = load_a64(p, ks * 32, hi);
#pragma unroll
      for (int t = 0; t < 4; ++t) acc[t] = wmma16(a, wf[ks][t], acc[t]);
    }
#pragma unroll
    for (int r = 0; r < 8; ++r) {
      long gr = (long)tile * 16 + 8 * hi + r;
      if (gr < N) {
#pragma unroll
        for (int t = 0; t < 4; ++t)
          out[gr * HDIM + t * 16 + l15] = acc[t][r];
      }
    }
  }
}

// ======== GCN scatter: agg[dst] += hw[src] * dinv[src]*dinv[dst] ========
__global__ void k_conv_scatter(const float* __restrict__ hw,
                               const int* __restrict__ src, const int* __restrict__ dst,
                               const float* __restrict__ dinv, float* agg, long E) {
  long tid = (long)blockIdx.x * blockDim.x + threadIdx.x;
  if (tid >= E * 16) return;
  long e = tid >> 4;
  int q = (int)(tid & 15);
  int s = src[e], d = dst[e];
  float c = dinv[s] * dinv[d];
  float4 v = *(const float4*)(hw + (long)s * HDIM + q * 4);
  float* ap = agg + (long)d * HDIM + q * 4;
  atomicAdd(ap + 0, v.x * c);
  atomicAdd(ap + 1, v.y * c);
  atomicAdd(ap + 2, v.z * c);
  atomicAdd(ap + 3, v.w * c);
}

// out = [relu]( agg + hw * dinv^2 + bias ), float4-wide
__global__ void k_conv_fin4(const float4* __restrict__ agg, const float4* __restrict__ hw,
                            const float* __restrict__ dinv, const float4* __restrict__ bias,
                            float4* out, long total4, int do_relu) {
  long tid = (long)blockIdx.x * blockDim.x + threadIdx.x;
  if (tid >= total4) return;
  long i = tid >> 4;          // 16 float4 per 64-wide row
  int f = (int)(tid & 15);
  float di = dinv[i];
  float c = di * di;
  float4 a = agg[tid], h = hw[tid], b = bias[f];
  float4 v = make_float4(a.x + h.x * c + b.x, a.y + h.y * c + b.y,
                         a.z + h.z * c + b.z, a.w + h.w * c + b.w);
  if (do_relu) {
    v.x = fmaxf(v.x, 0.f); v.y = fmaxf(v.y, 0.f);
    v.z = fmaxf(v.z, 0.f); v.w = fmaxf(v.w, 0.f);
  }
  out[tid] = v;
}

// ========== edge encoder MLP + pool-by-src (atomic scatter from C fragments) ==========
__global__ void k_edge_enc(const float* __restrict__ ea, const int* __restrict__ src,
                           const float* __restrict__ We1, const float* __restrict__ be1,
                           const float* __restrict__ We2, const float* __restrict__ be2,
                           float* epool, long E) {
  __shared__ float lds[WAVES][16 * LDSS];
  int wv = threadIdx.x >> 5, lane = threadIdx.x & 31;
  int hi = lane >> 4, l15 = lane & 15;
  long etiles = (E + 15) >> 4;
  float* L = lds[wv];

  v16h w1[4], w2[2][4];
  float b1c[4], b2c[4];
#pragma unroll
  for (int t = 0; t < 4; ++t) {
    int col = t * 16 + l15;
    w1[t] = load_bfrag(We1, 0, col, hi, 4);
    w2[0][t] = load_bfrag(We2, 0, col, hi, HDIM);
    w2[1][t] = load_bfrag(We2, 32, col, hi, HDIM);
    b1c[t] = be1[col];
    b2c[t] = be2[col];
  }

  long stride = (long)gridDim.x * WAVES;
  for (long tile = (long)blockIdx.x * WAVES + wv; tile < etiles; tile += stride) {
    long rowg = tile * 16 + l15;
    if (rowg >= E) rowg = E - 1;

    // layer 1: [16,4] @ [4,64] — edge row is exactly one float4
    float4 u = *(const float4*)(ea + rowg * 4);
    v16h a;
#pragma unroll
    for (int t = 0; t < 8; ++t) {
      float v = (t == 0) ? u.x : (t == 1) ? u.y : (t == 2) ? u.z : (t == 3) ? u.w : 0.0f;
      a[t]     = (_Float16)((hi == 0 && t < 4) ? v : 0.0f);
      a[8 + t] = (_Float16)0.0f;
    }
    v8f acc[4] = {};
#pragma unroll
    for (int t = 0; t < 4; ++t) acc[t] = wmma16(a, w1[t], acc[t]);

#pragma unroll
    for (int t = 0; t < 4; ++t) {
      int col = t * 16 + l15;
#pragma unroll
      for (int r = 0; r < 8; ++r)
        L[(8 * hi + r) * LDSS + col] = fmaxf(acc[t][r] + b1c[t], 0.0f);
    }
    __builtin_amdgcn_wave_barrier();

    // layer 2: [16,64] @ [64,64]
    v8f acc2[4] = {};
#pragma unroll
    for (int ks = 0; ks < 2; ++ks) {
      v16h a2 = load_a_lds(L, l15, ks * 32, hi);
#pragma unroll
      for (int t = 0; t < 4; ++t) acc2[t] = wmma16(a2, w2[ks][t], acc2[t]);
    }
    __builtin_amdgcn_wave_barrier();

    // scatter pooled sums: epool[src[e]] += e_feat
#pragma unroll
    for (int r = 0; r < 8; ++r) {
      long ge = tile * 16 + 8 * hi + r;
      if (ge < E) {
        int s = src[ge];
        float* pp = epool + (long)s * HDIM;
#pragma unroll
        for (int t = 0; t < 4; ++t)
          atomicAdd(pp + t * 16 + l15, acc2[t][r] + b2c[t]);
      }
    }
  }
}

__global__ void k_pool_div4(float4* epool, const float* __restrict__ scnt, long total4) {
  long tid = (long)blockIdx.x * blockDim.x + threadIdx.x;
  if (tid >= total4) return;
  long i = tid >> 4;
  float inv = 1.0f / fmaxf(scnt[i], 1.0f);
  float4 v = epool[tid];
  epool[tid] = make_float4(v.x * inv, v.y * inv, v.z * inv, v.w * inv);
}

// ========== head: relu([h2, epool] @ Wh1 + bh1) @ Wh2 + bh2 ==========
__global__ void k_head(const float* __restrict__ h2, const float* __restrict__ ep,
                       const float* __restrict__ Wh1, const float* __restrict__ bh1,
                       const float* __restrict__ Wh2, const float* __restrict__ bh2,
                       float* out, int N) {
  __shared__ float lds[WAVES][16 * LDSS];
  __shared__ float w2s[HDIM];
  int wv = threadIdx.x >> 5, lane = threadIdx.x & 31;
  int hi = lane >> 4, l15 = lane & 15;
  int ntiles = (N + 15) >> 4;
  float* L = lds[wv];

  if (threadIdx.x < HDIM) w2s[threadIdx.x] = Wh2[threadIdx.x];
  __syncthreads();   // uniform, before the grid-stride loop

  v16h wf[4][4];
  float b1c[4];
#pragma unroll
  for (int ks = 0; ks < 4; ++ks)
#pragma unroll
    for (int t = 0; t < 4; ++t)
      wf[ks][t] = load_bfrag(Wh1, ks * 32, t * 16 + l15, hi, 128);
#pragma unroll
  for (int t = 0; t < 4; ++t) b1c[t] = bh1[t * 16 + l15];
  float bout = bh2[0];

  int stride = gridDim.x * WAVES;
  for (int tile = blockIdx.x * WAVES + wv; tile < ntiles; tile += stride) {
    long rowg = (long)tile * 16 + l15;
    if (rowg >= N) rowg = N - 1;

    v8f acc[4] = {};
#pragma unroll
    for (int ks = 0; ks < 4; ++ks) {
      const float* As = (ks < 2) ? h2 : ep;
      v16h a = load_a64(As + rowg * HDIM, (ks & 1) * 32, hi);
#pragma unroll
      for (int t = 0; t < 4; ++t) acc[t] = wmma16(a, wf[ks][t], acc[t]);
    }
#pragma unroll
    for (int t = 0; t < 4; ++t) {
      int col = t * 16 + l15;
#pragma unroll
      for (int r = 0; r < 8; ++r)
        L[(8 * hi + r) * LDSS + col] = fmaxf(acc[t][r] + b1c[t], 0.0f);
    }
    __builtin_amdgcn_wave_barrier();

    // row dot with Wh2[64]: lane pair (l15, l15+16) splits the 64 terms
    float sum = 0.0f;
    const float* q = L + l15 * LDSS + hi * 32;
    const float* w = w2s + hi * 32;
#pragma unroll
    for (int f = 0; f < 32; ++f) sum += q[f] * w[f];
    sum += __shfl_xor(sum, 16, 32);
    __builtin_amdgcn_wave_barrier();
    long gr = (long)tile * 16 + l15;
    if (hi == 0 && gr < N) out[gr] = sum + bout;
  }
}

// ================= host side =================
extern "C" void kernel_launch(void* const* d_in, const int* in_sizes, int n_in,
                              void* d_out, int out_size, void* d_ws, size_t ws_size,
                              hipStream_t stream) {
  const float* x   = (const float*)d_in[0];
  const float* ea  = (const float*)d_in[1];
  const int*   ei  = (const int*)d_in[2];
  // d_in[3] = batch (arange -> mean_pool over nodes is identity), unused
  const float* Wn1 = (const float*)d_in[4];  const float* bn1 = (const float*)d_in[5];
  const float* Wn2 = (const float*)d_in[6];  const float* bn2 = (const float*)d_in[7];
  const float* Wc1 = (const float*)d_in[8];  const float* bc1 = (const float*)d_in[9];
  const float* Wc2 = (const float*)d_in[10]; const float* bc2 = (const float*)d_in[11];
  const float* We1 = (const float*)d_in[12]; const float* be1 = (const float*)d_in[13];
  const float* We2 = (const float*)d_in[14]; const float* be2 = (const float*)d_in[15];
  const float* Wh1 = (const float*)d_in[16]; const float* bh1 = (const float*)d_in[17];
  const float* Wh2 = (const float*)d_in[18]; const float* bh2 = (const float*)d_in[19];

  const int  N = in_sizes[0] / 5;
  const long E = in_sizes[1] / 4;
  const int* src = ei;
  const int* dst = ei + E;

  float* ws   = (float*)d_ws;
  float* bufA = ws;                       // h0 -> h2
  float* bufB = bufA + (long)N * HDIM;    // h1
  float* bufC = bufB + (long)N * HDIM;    // hw -> epool
  float* bufD = bufC + (long)N * HDIM;    // agg
  float* deg  = bufD + (long)N * HDIM;    // [N]
  float* dinv = deg + N;                  // [N]
  float* scnt = dinv + N;                 // [N]

  const long NH  = (long)N * HDIM;
  const long NH4 = NH / 4;
  const int ntiles  = (N + 15) / 16;
  int nodeblocks = (ntiles + WAVES - 1) / WAVES;
  if (nodeblocks > 512) nodeblocks = 512;
  const long etiles = (E + 15) / 16;
  int edgeblocks = (int)((etiles + WAVES - 1) / WAVES);
  if (edgeblocks > 2048) edgeblocks = 2048;

  auto zgrid = [](long n) { return dim3((unsigned)((n + 255) / 256)); };

  // degrees + dinv
  k_zero<<<zgrid(3L * N), 256, 0, stream>>>(deg, 3L * N);
  k_deg<<<zgrid(E), 256, 0, stream>>>(src, dst, deg, scnt, E);
  k_dinv<<<zgrid(N), 256, 0, stream>>>(deg, dinv, N);

  // node encoder
  k_node_enc<<<nodeblocks, TPB, 0, stream>>>(x, Wn1, bn1, Wn2, bn2, bufA, N);

  // GCN conv 1 (relu)
  k_zero4<<<zgrid(NH4), 256, 0, stream>>>((float4*)bufD, NH4);
  k_gemm64<<<nodeblocks, TPB, 0, stream>>>(bufA, Wc1, bufC, N);
  k_conv_scatter<<<zgrid(E * 16), 256, 0, stream>>>(bufC, src, dst, dinv, bufD, E);
  k_conv_fin4<<<zgrid(NH4), 256, 0, stream>>>((const float4*)bufD, (const float4*)bufC,
                                              dinv, (const float4*)bc1, (float4*)bufB, NH4, 1);

  // GCN conv 2 (no relu)
  k_zero4<<<zgrid(NH4), 256, 0, stream>>>((float4*)bufD, NH4);
  k_gemm64<<<nodeblocks, TPB, 0, stream>>>(bufB, Wc2, bufC, N);
  k_conv_scatter<<<zgrid(E * 16), 256, 0, stream>>>(bufC, src, dst, dinv, bufD, E);
  k_conv_fin4<<<zgrid(NH4), 256, 0, stream>>>((const float4*)bufD, (const float4*)bufC,
                                              dinv, (const float4*)bc2, (float4*)bufA, NH4, 0);

  // edge encoder + mean pool by src (reuse bufC as epool)
  k_zero4<<<zgrid(NH4), 256, 0, stream>>>((float4*)bufC, NH4);
  k_edge_enc<<<edgeblocks, TPB, 0, stream>>>(ea, src, We1, be1, We2, be2, bufC, E);
  k_pool_div4<<<zgrid(NH4), 256, 0, stream>>>((float4*)bufC, scnt, NH4);

  // prediction head
  k_head<<<nodeblocks, TPB, 0, stream>>>(bufA, bufC, Wh1, bh1, Wh2, bh2, (float*)d_out, N);
}
